// VariationalAutoEncoder_43035572306782
// MI455X (gfx1250) — compile-verified, matrix operand order checked by reading, not run on hardware
//
#include <hip/hip_runtime.h>
#include <hip/hip_bf16.h>

// ---------------------------------------------------------------------------
// LSTM-VAE for MI455X (gfx1250, wave32, WMMA).
// - All GEMMs via v_wmma_f32_16x16x32_bf16 (bf16 in, f32 accum).
// - One persistent kernel per LSTM layer: fuses xt@Wx + h@Wh + bias + gates,
//   one grid-wide barrier per timestep, ping-pong h buffers.
// - Cell state c kept entirely in registers (block owns its 16x16 patch for
//   all T steps) -> no cell-state memory traffic at all.
// - Per-block weight slices staged into LDS once via
//   global_load_async_to_lds_b128 (ASYNCcnt-tracked), reused for 256 steps;
//   inner-loop B operands come from ds_load_b128, chunk-major layout.
// ---------------------------------------------------------------------------

#define DEV __device__ __forceinline__

typedef __attribute__((ext_vector_type(16))) __bf16 v16bf;
typedef __attribute__((ext_vector_type(8)))  __bf16 v8bf;
typedef __attribute__((ext_vector_type(8)))  float  v8f;
typedef __attribute__((ext_vector_type(4)))  int    v4i;

// address-space-qualified b128 pointers for the async-to-LDS builtin:
// param 1 is AS1 (global), param 2 is AS3 (LDS)
typedef v4i __attribute__((address_space(1))) v4i_glb;
typedef v4i __attribute__((address_space(3))) v4i_lds;

#if defined(__has_builtin)
#if __has_builtin(__builtin_amdgcn_global_load_async_to_lds_b128) && \
    __has_builtin(__builtin_amdgcn_s_wait_asynccnt)
#define ASYNC_LDS 1
#endif
#endif

constexpr int B_ = 256;   // batch
constexpr int T_ = 256;   // time
constexpr int U_ = 512;   // units
constexpr int L_ = 256;   // latent
constexpr int V_ = 64;    // vocab
constexpr int G_ = 2048;  // 4*U gates

// 16x32 bf16 tile operand from global memory (CDNA5 16-bit layout):
// lane half selects K chunks {0..7,16..23} vs {8..15,24..31};
// caller's pointer already includes row base + half*8.
DEV v16bf load_tile16(const __bf16* p) {
  v8bf lo = *(const v8bf*)p;
  v8bf hi = *(const v8bf*)(p + 16);
  return __builtin_shufflevector(lo, hi, 0,1,2,3,4,5,6,7,8,9,10,11,12,13,14,15);
}

// Same operand from an LDS weight slice stored chunk-major:
// 16B chunk c of row r (k = c*8 .. c*8+7) lives at region[(c*64 + r)*8].
// 16 lanes * 16B span all 64 banks -> conflict-free within a half-wave.
DEV v16bf lds_tile16(const __bf16* region, int row, int kbase) {
  const int c = kbase >> 3;
  v8bf lo = *(const v8bf*)(region + ((size_t)(c)     * 64 + row) * 8);
  v8bf hi = *(const v8bf*)(region + ((size_t)(c + 2) * 64 + row) * 8);
  return __builtin_shufflevector(lo, hi, 0,1,2,3,4,5,6,7,8,9,10,11,12,13,14,15);
}

// One 16B global->LDS staging copy (async DMA path on CDNA5 toolchains).
DEV void stage16(const __bf16* g, __bf16* l) {
#ifdef ASYNC_LDS
  __builtin_amdgcn_global_load_async_to_lds_b128(
      (v4i_glb*)g, (v4i_lds*)l, 0, 0);
#else
  *(v8bf*)l = *(const v8bf*)g;
#endif
}

DEV void stage_fence() {
#ifdef ASYNC_LDS
  __builtin_amdgcn_s_wait_asynccnt(0);
#endif
  __syncthreads();
}

DEV float sigmoidf_(float x) { return 1.f / (1.f + __expf(-x)); }
DEV float tanhf_(float x)    { return 2.f / (1.f + __expf(-2.f * x)) - 1.f; }

DEV void grid_sync(unsigned* cnt, unsigned* gen, unsigned nb) {
  __threadfence();
  if (threadIdx.x == 0) {
    unsigned g = __hip_atomic_load(gen, __ATOMIC_ACQUIRE, __HIP_MEMORY_SCOPE_AGENT);
    unsigned a = __hip_atomic_fetch_add(cnt, 1u, __ATOMIC_ACQ_REL, __HIP_MEMORY_SCOPE_AGENT);
    if (a == nb - 1u) {
      __hip_atomic_store(cnt, 0u, __ATOMIC_RELAXED, __HIP_MEMORY_SCOPE_AGENT);
      __hip_atomic_fetch_add(gen, 1u, __ATOMIC_RELEASE, __HIP_MEMORY_SCOPE_AGENT);
    } else {
      while (__hip_atomic_load(gen, __ATOMIC_ACQUIRE, __HIP_MEMORY_SCOPE_AGENT) == g)
        __builtin_amdgcn_s_sleep(2);
    }
  }
  __syncthreads();
  __threadfence();
}

// ---- small elementwise kernels --------------------------------------------

__global__ void k_zero(unsigned* __restrict__ p, int n) {
  int i = blockIdx.x * blockDim.x + threadIdx.x;
  if (i < n) p[i] = 0u;
}

// dst[N,K] (bf16) = transpose of src[K,N] (f32)
__global__ void k_transpose_bf16(const float* __restrict__ src,
                                 __bf16* __restrict__ dst, int K, int N) {
  int idx = blockIdx.x * blockDim.x + threadIdx.x;
  if (idx >= K * N) return;
  int k = idx / N, n = idx - k * N;
  dst[(size_t)n * K + k] = (__bf16)src[idx];
}

__global__ void k_embed(const int* __restrict__ tok, const float* __restrict__ emb,
                        __bf16* __restrict__ x) {
  int idx = blockIdx.x * blockDim.x + threadIdx.x;   // B*T*V
  int v = idx & (V_ - 1);
  int bt = idx >> 6;
  x[idx] = (__bf16)emb[tok[bt] * V_ + v];
}

__global__ void k_decin(const __bf16* __restrict__ latent,
                        const __bf16* __restrict__ xemb,
                        __bf16* __restrict__ decin) {
  int idx = blockIdx.x * blockDim.x + threadIdx.x;   // B*T*(L+V)
  if (idx >= B_ * T_ * (L_ + V_)) return;
  int d = idx % (L_ + V_);
  int bt = idx / (L_ + V_);
  int b = bt >> 8;                                   // T == 256
  decin[idx] = (d < L_) ? latent[b * L_ + d] : xemb[(size_t)bt * V_ + (d - L_)];
}

__global__ void k_argmax(const float* __restrict__ logits, float* __restrict__ pred) {
  int i = blockIdx.x * blockDim.x + threadIdx.x;     // B*T
  const float* p = logits + (size_t)i * V_;
  float best = p[0];
  int bi = 0;
  for (int j = 1; j < V_; ++j) { float v = p[j]; if (v > best) { best = v; bi = j; } }
  pred[i] = (float)bi;
}

// ---- fused persistent LSTM layer ------------------------------------------
// grid = 512 one-wave blocks (16 batch-row tiles x 32 unit tiles). Each block
// owns a 16x16 (batch, units) patch: 4 gate tiles, A tile reused across 4
// WMMAs, c state in registers, Wx/Wh slices in LDS.

template<int DIN>
__global__ __launch_bounds__(32) void k_lstm(
    const __bf16* __restrict__ X,    // [B,T,DIN]
    const __bf16* __restrict__ WxT,  // [2048, DIN]  (transposed)
    const __bf16* __restrict__ WhT,  // [2048, 512]  (transposed)
    const float*  __restrict__ bias, // [2048]
    __bf16* __restrict__ Y,          // [B,T,512]
    __bf16* __restrict__ H0, __bf16* __restrict__ H1,  // ping-pong h (bf16)
    unsigned* __restrict__ barCnt, unsigned* __restrict__ barGen)
{
  extern __shared__ __bf16 smem[];
  __bf16* WxL = smem;                 // 64*DIN elems, chunk-major
  __bf16* WhL = smem + 64 * DIN;      // 64*512 elems, chunk-major

  const int lane = threadIdx.x;
  const int n    = lane & 15;
  const int half = lane >> 4;
  const int hoff = half * 8;
  const int m0 = (blockIdx.x & 15) << 4;   // batch-row tile
  const int u0 = (blockIdx.x >> 4) << 4;   // unit-col tile
  const unsigned nb = gridDim.x;

  // ---- stage this block's weight slices into LDS (once, reused 256x) ----
  const int xch = DIN >> 3;                // 16B chunks per Wx row
  for (int idx = lane; idx < 64 * xch; idx += 32) {
    const int c = idx >> 6, r = idx & 63;
    const int g = r >> 4, nn = r & 15;
    stage16(WxT + (size_t)(g * U_ + u0 + nn) * DIN + (size_t)c * 8,
            WxL + ((size_t)c * 64 + r) * 8);
  }
  for (int idx = lane; idx < 64 * 64; idx += 32) {
    const int c = idx >> 6, r = idx & 63;
    const int g = r >> 4, nn = r & 15;
    stage16(WhT + (size_t)(g * U_ + u0 + nn) * U_ + (size_t)c * 8,
            WhL + ((size_t)c * 64 + r) * 8);
  }
  stage_fence();

  const __bf16* xrow = X + (size_t)(m0 + n) * T_ * DIN + hoff;
  float bg4[4];
#pragma unroll
  for (int g = 0; g < 4; ++g) bg4[g] = bias[g * U_ + u0 + n];

  // cell state lives in registers: element (M = r + 8*half, N = lane&15)
  float creg[8];
#pragma unroll
  for (int r = 0; r < 8; ++r) creg[r] = 0.f;

  for (int t = 0; t < T_; ++t) {
    const __bf16* Hr = (t & 1) ? H1 : H0;
    __bf16*       Hw = (t & 1) ? H0 : H1;

    v8f acc[4] = {};

    // xt @ Wx  (K = DIN); A from global, B from LDS
    const __bf16* xa = xrow + (size_t)t * DIN;
    if (t + 1 < T_) __builtin_prefetch(xa + DIN, 0, 1);
#pragma unroll 2
    for (int k0 = 0; k0 < DIN; k0 += 32) {
      const v16bf a = load_tile16(xa + k0);
#pragma unroll
      for (int g = 0; g < 4; ++g) {
        const v16bf b = lds_tile16(WxL, g * 16 + n, k0 + hoff);
        acc[g] = __builtin_amdgcn_wmma_f32_16x16x32_bf16(
            false, a, false, b, (short)0, acc[g], false, false);
      }
    }

    // h @ Wh  (K = 512)
    const __bf16* ha = Hr + (size_t)(m0 + n) * U_ + hoff;
#pragma unroll 2
    for (int k0 = 0; k0 < U_; k0 += 32) {
      const v16bf a = load_tile16(ha + k0);
#pragma unroll
      for (int g = 0; g < 4; ++g) {
        const v16bf b = lds_tile16(WhL, g * 16 + n, k0 + hoff);
        acc[g] = __builtin_amdgcn_wmma_f32_16x16x32_bf16(
            false, a, false, b, (short)0, acc[g], false, false);
      }
    }

    // gates + state update
#pragma unroll
    for (int r = 0; r < 8; ++r) {
      const int row = m0 + r + 8 * half;
      const float iv = sigmoidf_(acc[0][r] + bg4[0]);
      const float fv = sigmoidf_(acc[1][r] + bg4[1]);
      const float gv = tanhf_   (acc[2][r] + bg4[2]);
      const float ov = sigmoidf_(acc[3][r] + bg4[3]);
      const float c  = fv * creg[r] + iv * gv;
      creg[r] = c;
      const float h = ov * tanhf_(c);
      const __bf16 hb = (__bf16)h;
      Hw[row * U_ + u0 + n] = hb;
      Y[((size_t)row * T_ + t) * U_ + u0 + n] = hb;
    }

    grid_sync(barCnt, barGen, nb);
  }
}

// ---- latent heads: mean/sigma GEMMs + reparameterization -------------------
__global__ __launch_bounds__(32) void k_latent(
    const __bf16* __restrict__ Hf,   // [256,512] final encoder h (bf16)
    const __bf16* __restrict__ WmT,  // [256,512]
    const __bf16* __restrict__ WsT,  // [256,512]
    const float* __restrict__ bm, const float* __restrict__ bs,
    const float* __restrict__ eps,   // [256,256]
    __bf16* __restrict__ latent)     // [256,256] bf16
{
  const int lane = threadIdx.x;
  const int n = lane & 15, half = lane >> 4, hoff = half * 8;
  const int m0 = (blockIdx.x & 15) << 4;
  const int u0 = (blockIdx.x >> 4) << 4;

  const __bf16* ha = Hf  + (size_t)(m0 + n) * U_ + hoff;
  const __bf16* wm = WmT + (size_t)(u0 + n) * U_ + hoff;
  const __bf16* ws = WsT + (size_t)(u0 + n) * U_ + hoff;

  v8f am = {};
  v8f as = {};
#pragma unroll 4
  for (int k0 = 0; k0 < U_; k0 += 32) {
    const v16bf a  = load_tile16(ha + k0);
    const v16bf b0 = load_tile16(wm + k0);
    const v16bf b1 = load_tile16(ws + k0);
    am = __builtin_amdgcn_wmma_f32_16x16x32_bf16(false, a, false, b0, (short)0, am, false, false);
    as = __builtin_amdgcn_wmma_f32_16x16x32_bf16(false, a, false, b1, (short)0, as, false, false);
  }
  const float bm0 = bm[u0 + n], bs0 = bs[u0 + n];
#pragma unroll
  for (int r = 0; r < 8; ++r) {
    const int row = m0 + r + 8 * half, col = u0 + n;
    const float mean = am[r] + bm0;
    const float sg   = as[r] + bs0;
    latent[row * L_ + col] = (__bf16)(mean + __expf(0.5f * sg) * eps[row * L_ + col]);
  }
}

// ---- output projection: logits = Y @ dec_W + b -----------------------------
__global__ __launch_bounds__(32) void k_proj(
    const __bf16* __restrict__ Yseq, // [B*T, 512]
    const __bf16* __restrict__ WdT,  // [64, 512]
    const float* __restrict__ db,    // [64]
    float* __restrict__ logits)      // [B*T, 64]
{
  const int lane = threadIdx.x;
  const int n = lane & 15, half = lane >> 4, hoff = half * 8;
  const int m0 = blockIdx.x << 4;    // 4096 row tiles

  const __bf16* ar = Yseq + (size_t)(m0 + n) * U_ + hoff;
  const __bf16* br[4];
#pragma unroll
  for (int j = 0; j < 4; ++j) br[j] = WdT + (size_t)(j * 16 + n) * U_ + hoff;

  v8f acc[4] = {};
#pragma unroll 2
  for (int k0 = 0; k0 < U_; k0 += 32) {
    const v16bf a = load_tile16(ar + k0);
#pragma unroll
    for (int j = 0; j < 4; ++j) {
      const v16bf b = load_tile16(br[j] + k0);
      acc[j] = __builtin_amdgcn_wmma_f32_16x16x32_bf16(
          false, a, false, b, (short)0, acc[j], false, false);
    }
  }
#pragma unroll
  for (int j = 0; j < 4; ++j) {
    const int col = j * 16 + n;
    const float bb = db[col];
#pragma unroll
    for (int r = 0; r < 8; ++r) {
      const int row = m0 + r + 8 * half;
      logits[(size_t)row * V_ + col] = acc[j][r] + bb;
    }
  }
}

// ---------------------------------------------------------------------------

extern "C" void kernel_launch(void* const* d_in, const int* in_sizes, int n_in,
                              void* d_out, int out_size, void* d_ws, size_t ws_size,
                              hipStream_t stream) {
  (void)in_sizes; (void)n_in; (void)out_size; (void)ws_size;
  const int*   tokens   = (const int*)  d_in[0];
  const float* eps      = (const float*)d_in[1];
  const float* emb      = (const float*)d_in[2];
  const float* enc_Wx0  = (const float*)d_in[3];
  const float* enc_Wh0  = (const float*)d_in[4];
  const float* enc_b0   = (const float*)d_in[5];
  const float* enc_Wx12 = (const float*)d_in[6];
  const float* enc_Wh12 = (const float*)d_in[7];
  const float* enc_b12  = (const float*)d_in[8];
  const float* w_mean   = (const float*)d_in[9];
  const float* b_mean   = (const float*)d_in[10];
  const float* w_sigma  = (const float*)d_in[11];
  const float* b_sigma  = (const float*)d_in[12];
  const float* dec_Wx0  = (const float*)d_in[13];
  const float* dec_Wh0  = (const float*)d_in[14];
  const float* dec_b0   = (const float*)d_in[15];
  const float* dec_Wx12 = (const float*)d_in[16];
  const float* dec_Wh12 = (const float*)d_in[17];
  const float* dec_b12  = (const float*)d_in[18];
  const float* dec_W    = (const float*)d_in[19];
  const float* dec_b    = (const float*)d_in[20];

  char* base = (char*)d_ws;
  size_t off = 0;
  auto alloc = [&](size_t bytes) -> void* {
    void* p = base + off;
    off = (off + bytes + 255) & ~(size_t)255;
    return p;
  };

  // transposed bf16 weights
  __bf16* WxT_e0 = (__bf16*)alloc((size_t)G_ * V_ * 2);
  __bf16* WhT_e0 = (__bf16*)alloc((size_t)G_ * U_ * 2);
  __bf16* WxT_e1 = (__bf16*)alloc((size_t)G_ * U_ * 2);
  __bf16* WhT_e1 = (__bf16*)alloc((size_t)G_ * U_ * 2);
  __bf16* WxT_e2 = (__bf16*)alloc((size_t)G_ * U_ * 2);
  __bf16* WhT_e2 = (__bf16*)alloc((size_t)G_ * U_ * 2);
  __bf16* WxT_d0 = (__bf16*)alloc((size_t)G_ * (L_ + V_) * 2);
  __bf16* WhT_d0 = (__bf16*)alloc((size_t)G_ * U_ * 2);
  __bf16* WxT_d1 = (__bf16*)alloc((size_t)G_ * U_ * 2);
  __bf16* WhT_d1 = (__bf16*)alloc((size_t)G_ * U_ * 2);
  __bf16* WxT_d2 = (__bf16*)alloc((size_t)G_ * U_ * 2);
  __bf16* WhT_d2 = (__bf16*)alloc((size_t)G_ * U_ * 2);
  __bf16* WmT    = (__bf16*)alloc((size_t)L_ * U_ * 2);
  __bf16* WsT    = (__bf16*)alloc((size_t)L_ * U_ * 2);
  __bf16* WdT    = (__bf16*)alloc((size_t)V_ * U_ * 2);
  // activations
  __bf16* xemb   = (__bf16*)alloc((size_t)B_ * T_ * V_ * 2);
  __bf16* seqA   = (__bf16*)alloc((size_t)B_ * T_ * U_ * 2);
  __bf16* seqB   = (__bf16*)alloc((size_t)B_ * T_ * U_ * 2);
  __bf16* decin  = (__bf16*)alloc((size_t)B_ * T_ * (L_ + V_) * 2);
  __bf16* latent = (__bf16*)alloc((size_t)B_ * L_ * 2);
  // zero-initialized recurrent state region (contiguous: H0|H1|bar)
  char* zbase = base + off;
  __bf16*   H0  = (__bf16*)alloc((size_t)B_ * U_ * 2);
  __bf16*   H1  = (__bf16*)alloc((size_t)B_ * U_ * 2);
  unsigned* bar = (unsigned*)alloc(256);
  const size_t zbytes = (size_t)((base + off) - zbase);

  auto xpose = [&](const float* src, __bf16* dst, int K, int N) {
    int total = K * N;
    k_transpose_bf16<<<(total + 255) / 256, 256, 0, stream>>>(src, dst, K, N);
  };
  auto zero_state = [&]() {
    int nw = (int)(zbytes / 4);
    k_zero<<<(nw + 255) / 256, 256, 0, stream>>>((unsigned*)zbase, nw);
  };

  // dynamic-LDS sizes per layer (chunk-major slices: 64*(DIN+512) bf16)
  const int sh64  = 64 * (V_       + U_) * 2;   //  73,728 B
  const int sh320 = 64 * (L_ + V_  + U_) * 2;   // 106,496 B
  const int sh512 = 64 * (U_       + U_) * 2;   // 131,072 B
  (void)hipFuncSetAttribute((const void*)&k_lstm<V_>,
      hipFuncAttributeMaxDynamicSharedMemorySize, sh64);
  (void)hipFuncSetAttribute((const void*)&k_lstm<L_ + V_>,
      hipFuncAttributeMaxDynamicSharedMemorySize, sh320);
  (void)hipFuncSetAttribute((const void*)&k_lstm<U_>,
      hipFuncAttributeMaxDynamicSharedMemorySize, sh512);

  // weight conversion + transpose (once per call)
  xpose(enc_Wx0, WxT_e0, V_, G_);
  xpose(enc_Wh0, WhT_e0, U_, G_);
  xpose(enc_Wx12 + 0 * (size_t)U_ * G_, WxT_e1, U_, G_);
  xpose(enc_Wx12 + 1 * (size_t)U_ * G_, WxT_e2, U_, G_);
  xpose(enc_Wh12 + 0 * (size_t)U_ * G_, WhT_e1, U_, G_);
  xpose(enc_Wh12 + 1 * (size_t)U_ * G_, WhT_e2, U_, G_);
  xpose(dec_Wx0, WxT_d0, L_ + V_, G_);
  xpose(dec_Wh0, WhT_d0, U_, G_);
  xpose(dec_Wx12 + 0 * (size_t)U_ * G_, WxT_d1, U_, G_);
  xpose(dec_Wx12 + 1 * (size_t)U_ * G_, WxT_d2, U_, G_);
  xpose(dec_Wh12 + 0 * (size_t)U_ * G_, WhT_d1, U_, G_);
  xpose(dec_Wh12 + 1 * (size_t)U_ * G_, WhT_d2, U_, G_);
  xpose(w_mean,  WmT, U_, L_);
  xpose(w_sigma, WsT, U_, L_);
  xpose(dec_W,   WdT, U_, V_);

  // embedding lookup -> bf16
  k_embed<<<(B_ * T_ * V_) / 256, 256, 0, stream>>>(tokens, emb, xemb);

  // encoder stack
  zero_state();
  k_lstm<V_><<<512, 32, sh64, stream>>>(xemb, WxT_e0, WhT_e0, enc_b0, seqA,
                                        H0, H1, bar, bar + 1);
  zero_state();
  k_lstm<U_><<<512, 32, sh512, stream>>>(seqA, WxT_e1, WhT_e1, enc_b12, seqB,
                                         H0, H1, bar, bar + 1);
  zero_state();
  k_lstm<U_><<<512, 32, sh512, stream>>>(seqB, WxT_e2, WhT_e2, enc_b12 + G_, seqA,
                                         H0, H1, bar, bar + 1);
  // T even -> final encoder h lives in H0

  // latent heads + reparameterization
  k_latent<<<256, 32, 0, stream>>>(H0, WmT, WsT, b_mean, b_sigma, eps, latent);

  // decoder input = concat(latent tiled over T, xemb)
  k_decin<<<(B_ * T_ * (L_ + V_) + 255) / 256, 256, 0, stream>>>(latent, xemb, decin);

  // decoder stack
  zero_state();
  k_lstm<L_ + V_><<<512, 32, sh320, stream>>>(decin, WxT_d0, WhT_d0, dec_b0, seqB,
                                              H0, H1, bar, bar + 1);
  zero_state();
  k_lstm<U_><<<512, 32, sh512, stream>>>(seqB, WxT_d1, WhT_d1, dec_b12, seqA,
                                         H0, H1, bar, bar + 1);
  zero_state();
  k_lstm<U_><<<512, 32, sh512, stream>>>(seqA, WxT_d2, WhT_d2, dec_b12 + G_, seqB,
                                         H0, H1, bar, bar + 1);

  // outputs: (pred [B*T], logits [B*T*V]) flattened
  float* outF   = (float*)d_out;
  float* logits = outF + (size_t)B_ * T_;
  k_proj<<<(B_ * T_) / 16, 32, 0, stream>>>(seqB, WdT, dec_b, logits);
  k_argmax<<<(B_ * T_) / 256, 256, 0, stream>>>(logits, outF);
}
